// ContentGuidedAttention_46213848105562
// MI455X (gfx1250) — compile-verified, matrix operand order checked by reading, not run on hardware
//
#include <hip/hip_runtime.h>
#include <hip/hip_bf16.h>

typedef __attribute__((ext_vector_type(16))) __bf16 v16bf;
typedef __attribute__((ext_vector_type(8)))  __bf16 v8bf;
typedef __attribute__((ext_vector_type(8)))  float  v8f;

static __device__ inline v8f v8f_zero() {
    v8f z;
#pragma unroll
    for (int i = 0; i < 8; ++i) z[i] = 0.0f;
    return z;
}

static __device__ inline v8f wmma_bf16(v16bf a, v16bf b, v8f c) {
    // D = A(16x32 bf16) * B(32x16 bf16) + C(16x16 f32)
    return __builtin_amdgcn_wmma_f32_16x16x32_bf16(
        /*neg_a=*/false, a, /*neg_b=*/false, b,
        /*c_mod=*/(short)0, c, /*reuse_a=*/false, /*reuse_b=*/false);
}

// Load a 16x32 bf16 operand tile (A layout; identical pattern serves B when the
// B source is stored "n-major" [n][k]).  Source is row-major with leading dim
// `ld` (elements).  Per CDNA5 ISA 16-bit A layout:
//   lanes 0-15 : row = rbase+lane,    K = {0..7, 16..23} + kbase
//   lanes 16-31: row = rbase+lane-16, K = {8..15, 24..31} + kbase
static __device__ inline v16bf load_tile(const __bf16* __restrict__ base, int ld,
                                         int rbase, int kbase, int lane) {
    const int r  = rbase + (lane & 15);
    const int k0 = kbase + ((lane & 16) ? 8 : 0);
    const __bf16* p = base + (long)r * ld + k0;
    union { v16bf v; v8bf h[2]; } u;
    u.h[0] = *reinterpret_cast<const v8bf*>(p);        // K = k0 .. k0+7
    u.h[1] = *reinterpret_cast<const v8bf*>(p + 16);   // K = k0+16 .. k0+23
    return u.v;
}

// ---------------------------------------------------------------------------
// Packing kernels
// ---------------------------------------------------------------------------
__global__ void pack_w4(const float* __restrict__ qw, const float* __restrict__ kw,
                        const float* __restrict__ vw, const float* __restrict__ ow,
                        __bf16* __restrict__ wq, __bf16* __restrict__ wk,
                        __bf16* __restrict__ wv, __bf16* __restrict__ wo) {
    int i = blockIdx.x * 256 + threadIdx.x;   // 65536 elements per matrix
    wq[i] = (__bf16)qw[i];
    wk[i] = (__bf16)kw[i];
    wv[i] = (__bf16)vw[i];
    wo[i] = (__bf16)ow[i];
}

// s3 [2][256][4096] f32  ->  x3t [2][4096][256] bf16
__global__ void pack_x3(const float* __restrict__ s3, __bf16* __restrict__ x3t) {
    int i = blockIdx.x * 256 + threadIdx.x;   // 2*256*4096 total
    int n = i & 4095;
    int c = (i >> 12) & 255;
    int b = i >> 20;
    x3t[((long)b * 4096 + n) * 256 + c] = (__bf16)s3[i];
}

// concat(s4 [2][256][2048], s5 [2][256][1024]) -> x45t [2][3072][256] bf16
__global__ void pack_x45(const float* __restrict__ s4, const float* __restrict__ s5,
                         __bf16* __restrict__ x45t) {
    int i = blockIdx.x * 256 + threadIdx.x;   // 2*256*3072 total
    int n = i % 3072;
    int c = (i / 3072) & 255;
    int b = i / (3072 * 256);
    float v = (n < 2048) ? s4[((long)b * 256 + c) * 2048 + n]
                         : s5[((long)b * 256 + c) * 1024 + (n - 2048)];
    x45t[((long)b * 3072 + n) * 256 + c] = (__bf16)v;
}

// ---------------------------------------------------------------------------
// Projection GEMM: Y[oc,n] = sum_c W[oc,c] * Xt[n][c] + bias[oc]
// 256 threads = 8 waves; wave w -> oc rows [32w, 32w+32); 64 n-columns per WG.
// transpose_out=1 -> store Y as [n][oc] bf16 (for Q/K); else [oc][n] (for V).
// ---------------------------------------------------------------------------
__global__ __launch_bounds__(256) void proj_gemm(const __bf16* __restrict__ Xt,
                                                 const __bf16* __restrict__ W,
                                                 const float* __restrict__ bias,
                                                 __bf16* __restrict__ Y,
                                                 int N, int transpose_out) {
    const int tilesN = N >> 6;
    const int b      = blockIdx.x / tilesN;
    const int nbase  = (blockIdx.x % tilesN) << 6;
    const int wave   = threadIdx.x >> 5;
    const int lane   = threadIdx.x & 31;
    const int ocbase = wave << 5;
    const __bf16* Xb = Xt + (long)b * N * 256;

    v8f acc[2][4];
#pragma unroll
    for (int i = 0; i < 2; ++i)
#pragma unroll
        for (int j = 0; j < 4; ++j) acc[i][j] = v8f_zero();

    for (int kk = 0; kk < 256; kk += 32) {
        v16bf a0 = load_tile(W, 256, ocbase,      kk, lane);
        v16bf a1 = load_tile(W, 256, ocbase + 16, kk, lane);
#pragma unroll
        for (int j = 0; j < 4; ++j) {
            v16bf bj = load_tile(Xb, 256, nbase + j * 16, kk, lane);
            acc[0][j] = wmma_bf16(a0, bj, acc[0][j]);
            acc[1][j] = wmma_bf16(a1, bj, acc[1][j]);
        }
    }

    const int hi = (lane & 16) ? 8 : 0;
#pragma unroll
    for (int i = 0; i < 2; ++i) {
        const int oc0 = ocbase + i * 16 + hi;
#pragma unroll
        for (int j = 0; j < 4; ++j) {
            const int n = nbase + j * 16 + (lane & 15);
            if (transpose_out) {
                v8bf pk;
#pragma unroll
                for (int r = 0; r < 8; ++r)
                    pk[r] = (__bf16)(acc[i][j][r] + bias[oc0 + r]);
                *reinterpret_cast<v8bf*>(Y + ((long)b * N + n) * 256 + oc0) = pk;
            } else {
#pragma unroll
                for (int r = 0; r < 8; ++r)
                    Y[((long)b * 256 + oc0 + r) * N + n] =
                        (__bf16)(acc[i][j][r] + bias[oc0 + r]);
            }
        }
    }
}

// ---------------------------------------------------------------------------
// Flash attention.  Grid: 512 WGs.  WG -> (b,h, 128-query block); each of the
// 8 waves owns a 16-query tile and streams Nk=3072 in 32-key chunks.
// Qt/Kt: [b][n][256] bf16 (n-major);  Vm: [b][256][3072] bf16 (c-major).
// Output AttnT: [b][q][256] bf16 (q-major, ready to be the next GEMM's B).
// ---------------------------------------------------------------------------
__global__ __launch_bounds__(256) void attn_kernel(const __bf16* __restrict__ Qt,
                                                   const __bf16* __restrict__ Kt,
                                                   const __bf16* __restrict__ Vm,
                                                   __bf16* __restrict__ AttnT) {
    __shared__ __bf16 P[8][16 * 32];   // per-wave probability staging tile

    const int wave  = threadIdx.x >> 5;
    const int lane  = threadIdx.x & 31;
    const int bh    = blockIdx.x >> 5;       // 0..15
    const int b     = bh >> 3;
    const int h     = bh & 7;
    const int qbase = ((blockIdx.x & 31) << 7) + (wave << 4);
    const int c0    = h * 32;

    const __bf16* Qb = Qt + (long)b * 4096 * 256;
    const __bf16* Kb = Kt + (long)b * 3072 * 256;
    const __bf16* Vb = Vm + (long)b * 256 * 3072;

    const v16bf aq = load_tile(Qb, 256, qbase, c0, lane);

    v8f o0 = v8f_zero(), o1 = v8f_zero();
    float mrun[8], lrun[8];
#pragma unroll
    for (int r = 0; r < 8; ++r) { mrun[r] = -1e30f; lrun[r] = 0.0f; }

    const float cexp = 0.25505402f;    // (1/sqrt(32)) * log2(e)
    __bf16* Pw = &P[wave][0];
    const int row = (lane & 16) ? 8 : 0;
    const int col = lane & 15;

    for (int kb = 0; kb < 3072; kb += 32) {
        v16bf bk0 = load_tile(Kb, 256, kb,      c0, lane);
        v16bf bk1 = load_tile(Kb, 256, kb + 16, c0, lane);
        v8f s0 = wmma_bf16(aq, bk0, v8f_zero());
        v8f s1 = wmma_bf16(aq, bk1, v8f_zero());

        float p0[8], p1[8];
#pragma unroll
        for (int r = 0; r < 8; ++r) {
            float x0 = s0[r] * cexp, x1 = s1[r] * cexp;
            float cm = fmaxf(x0, x1);
            cm = fmaxf(cm, __shfl_xor(cm, 1, 32));
            cm = fmaxf(cm, __shfl_xor(cm, 2, 32));
            cm = fmaxf(cm, __shfl_xor(cm, 4, 32));
            cm = fmaxf(cm, __shfl_xor(cm, 8, 32));
            float mnew = fmaxf(mrun[r], cm);
            float corr = exp2f(mrun[r] - mnew);
            p0[r] = exp2f(x0 - mnew);
            p1[r] = exp2f(x1 - mnew);
            float rs = p0[r] + p1[r];
            rs += __shfl_xor(rs, 1, 32);
            rs += __shfl_xor(rs, 2, 32);
            rs += __shfl_xor(rs, 4, 32);
            rs += __shfl_xor(rs, 8, 32);
            lrun[r] = lrun[r] * corr + rs;
            mrun[r] = mnew;
            o0[r] *= corr;
            o1[r] *= corr;
        }

        // D layout (lane=key, vgpr=query) -> row-major 16x32 in LDS
#pragma unroll
        for (int r = 0; r < 8; ++r) {
            Pw[(row + r) * 32 + col]      = (__bf16)p0[r];
            Pw[(row + r) * 32 + 16 + col] = (__bf16)p1[r];
        }
        v16bf ap  = load_tile(Pw, 32, 0, 0, lane);            // A layout read-back
        v16bf bv0 = load_tile(Vb, 3072, c0,      kb, lane);   // 32 keys x d 0..15
        v16bf bv1 = load_tile(Vb, 3072, c0 + 16, kb, lane);   // 32 keys x d 16..31
        o0 = wmma_bf16(ap, bv0, o0);
        o1 = wmma_bf16(ap, bv1, o1);
    }

#pragma unroll
    for (int r = 0; r < 8; ++r) {
        float inv = 1.0f / lrun[r];
        int q = qbase + row + r;
        AttnT[((long)b * 4096 + q) * 256 + c0 + col]      = (__bf16)(o0[r] * inv);
        AttnT[((long)b * 4096 + q) * 256 + c0 + 16 + col] = (__bf16)(o1[r] * inv);
    }
}

// ---------------------------------------------------------------------------
// Out-projection + bias + residual + LayerNorm(C) -> out [b][c][n] f32.
// WG = 256 threads, 32 n-columns; full 256 oc rows staged in 32 KB LDS.
// ---------------------------------------------------------------------------
__global__ __launch_bounds__(256) void outproj_ln(const __bf16* __restrict__ AttnT,
                                                  const __bf16* __restrict__ Wo,
                                                  const float* __restrict__ ob,
                                                  const float* __restrict__ s3,
                                                  const float* __restrict__ lnw,
                                                  const float* __restrict__ lnb,
                                                  float* __restrict__ out) {
    __shared__ float tile[256 * 32];
    __shared__ float red[2][8][32];
    __shared__ float mu_s[32], rs_s[32];

    const int tilesN = 4096 / 32;   // 128
    const int b      = blockIdx.x / tilesN;
    const int nbase  = (blockIdx.x % tilesN) * 32;
    const int wave   = threadIdx.x >> 5;
    const int lane   = threadIdx.x & 31;
    const int ocbase = wave << 5;
    const __bf16* Ab = AttnT + (long)b * 4096 * 256;

    v8f acc[2][2];
#pragma unroll
    for (int i = 0; i < 2; ++i)
#pragma unroll
        for (int j = 0; j < 2; ++j) acc[i][j] = v8f_zero();

    for (int kk = 0; kk < 256; kk += 32) {
        v16bf a0 = load_tile(Wo, 256, ocbase,      kk, lane);
        v16bf a1 = load_tile(Wo, 256, ocbase + 16, kk, lane);
#pragma unroll
        for (int j = 0; j < 2; ++j) {
            v16bf bj = load_tile(Ab, 256, nbase + j * 16, kk, lane);
            acc[0][j] = wmma_bf16(a0, bj, acc[0][j]);
            acc[1][j] = wmma_bf16(a1, bj, acc[1][j]);
        }
    }

    const int hi = (lane & 16) ? 8 : 0;
#pragma unroll
    for (int i = 0; i < 2; ++i) {
        const int oc0 = ocbase + i * 16 + hi;
#pragma unroll
        for (int j = 0; j < 2; ++j) {
            const int n = j * 16 + (lane & 15);
#pragma unroll
            for (int r = 0; r < 8; ++r) {
                float v = acc[i][j][r] + ob[oc0 + r] +
                          s3[((long)b * 256 + oc0 + r) * 4096 + nbase + n];
                tile[(oc0 + r) * 32 + n] = v;
            }
        }
    }
    __syncthreads();

    {   // partial sums: 8 channel-segments x 32 columns
        int ncol = threadIdx.x & 31, seg = threadIdx.x >> 5;
        float s = 0.f, sq = 0.f;
        for (int c = seg * 32; c < seg * 32 + 32; ++c) {
            float v = tile[c * 32 + ncol];
            s += v; sq += v * v;
        }
        red[0][seg][ncol] = s;
        red[1][seg][ncol] = sq;
    }
    __syncthreads();
    if (threadIdx.x < 32) {
        float s = 0.f, sq = 0.f;
        for (int g = 0; g < 8; ++g) { s += red[0][g][threadIdx.x]; sq += red[1][g][threadIdx.x]; }
        float mu  = s * (1.0f / 256.0f);
        float var = sq * (1.0f / 256.0f) - mu * mu;
        mu_s[threadIdx.x] = mu;
        rs_s[threadIdx.x] = rsqrtf(var + 1e-5f);
    }
    __syncthreads();
    {
        int ncol = threadIdx.x & 31, c0r = threadIdx.x >> 5;
        float mu = mu_s[ncol], rstd = rs_s[ncol];
        for (int c = c0r; c < 256; c += 8)
            out[((long)b * 256 + c) * 4096 + nbase + ncol] =
                (tile[c * 32 + ncol] - mu) * rstd * lnw[c] + lnb[c];
    }
}

// ---------------------------------------------------------------------------
extern "C" void kernel_launch(void* const* d_in, const int* in_sizes, int n_in,
                              void* d_out, int out_size, void* d_ws, size_t ws_size,
                              hipStream_t stream) {
    (void)in_sizes; (void)n_in; (void)out_size; (void)ws_size;

    const float* s3  = (const float*)d_in[0];
    const float* s4  = (const float*)d_in[1];
    const float* s5  = (const float*)d_in[2];
    const float* qw  = (const float*)d_in[3];
    const float* qb  = (const float*)d_in[4];
    const float* kw  = (const float*)d_in[5];
    const float* kb  = (const float*)d_in[6];
    const float* vw  = (const float*)d_in[7];
    const float* vb  = (const float*)d_in[8];
    const float* ow  = (const float*)d_in[9];
    const float* ob  = (const float*)d_in[10];
    const float* lnw = (const float*)d_in[11];
    const float* lnb = (const float*)d_in[12];
    float* out = (float*)d_out;

    char* ws = (char*)d_ws;
    size_t off = 0;
    auto alloc = [&](size_t bytes) { char* p = ws + off; off += (bytes + 255) & ~(size_t)255; return p; };

    __bf16* Wq    = (__bf16*)alloc(65536 * 2);
    __bf16* Wk    = (__bf16*)alloc(65536 * 2);
    __bf16* Wv    = (__bf16*)alloc(65536 * 2);
    __bf16* Wo    = (__bf16*)alloc(65536 * 2);
    __bf16* X3t   = (__bf16*)alloc((size_t)2 * 4096 * 256 * 2);
    __bf16* X45t  = (__bf16*)alloc((size_t)2 * 3072 * 256 * 2);
    __bf16* Qt    = (__bf16*)alloc((size_t)2 * 4096 * 256 * 2);
    __bf16* Kt    = (__bf16*)alloc((size_t)2 * 3072 * 256 * 2);
    __bf16* Vm    = (__bf16*)alloc((size_t)2 * 256 * 3072 * 2);
    __bf16* AttnT = (__bf16*)alloc((size_t)2 * 4096 * 256 * 2);

    pack_w4 <<<256,  256, 0, stream>>>(qw, kw, vw, ow, Wq, Wk, Wv, Wo);
    pack_x3 <<<8192, 256, 0, stream>>>(s3, X3t);
    pack_x45<<<6144, 256, 0, stream>>>(s4, s5, X45t);

    proj_gemm<<<128, 256, 0, stream>>>(X3t,  Wq, qb, Qt, 4096, 1);  // Q -> [n][c]
    proj_gemm<<< 96, 256, 0, stream>>>(X45t, Wk, kb, Kt, 3072, 1);  // K -> [n][c]
    proj_gemm<<< 96, 256, 0, stream>>>(X45t, Wv, vb, Vm, 3072, 0);  // V -> [c][n]

    attn_kernel<<<512, 256, 0, stream>>>(Qt, Kt, Vm, AttnT);

    outproj_ln<<<256, 256, 0, stream>>>(AttnT, Wo, ob, s3, lnw, lnb, out);
}
